// CrossAttention_36447092474067
// MI455X (gfx1250) — matrix operand order repaired and checked by reading, not compiled, not run on hardware
//
#include <hip/hip_runtime.h>

// ---------------- types ----------------
typedef __attribute__((ext_vector_type(16))) __bf16 v16bf;
typedef __attribute__((ext_vector_type(8)))  __bf16 v8bf;
typedef __attribute__((ext_vector_type(8)))  float  v8f;

#define NQ   2048
#define NKV  2048
#define DX   512
#define DCTX 768
#define DI   512
#define NH   8
#define DH   64

// ---------------- fragment loaders (ISA 7.12.2 layouts) ----------------
// A 16x32 bf16: lane L holds row M = L%16.
//   lanes 0-15 : elems 0..7 = K 0..7,   elems 8..15 = K 16..23
//   lanes 16-31: elems 0..7 = K 8..15,  elems 8..15 = K 24..31
__device__ __forceinline__ v16bf load_afrag(const __bf16* tile, int ld) {
  int lane = threadIdx.x & 31;
  int m = lane & 15, hf = lane >> 4;
  const __bf16* p = tile + (size_t)m * ld + hf * 8;
  v8bf lo = *(const v8bf*)(p);
  v8bf hi = *(const v8bf*)(p + 16);
  v16bf r;
#pragma unroll
  for (int i = 0; i < 8; ++i) { r[i] = lo[i]; r[i + 8] = hi[i]; }
  return r;
}

// B 32x16 bf16: lane L holds column N = L%16.
//   lanes 0-15 : elems 0..7 = K 0..7,   elems 8..15 = K 16..23
//   lanes 16-31: elems 0..7 = K 8..15,  elems 8..15 = K 24..31
// Bt must be stored [N][K] row-major (K contiguous per output column).
__device__ __forceinline__ v16bf load_bfrag(const __bf16* tile, int ld) {
  int lane = threadIdx.x & 31;
  int n = lane & 15, hf = lane >> 4;
  const __bf16* p = tile + (size_t)n * ld + hf * 8;
  v8bf lo = *(const v8bf*)(p);
  v8bf hi = *(const v8bf*)(p + 16);
  v16bf r;
#pragma unroll
  for (int i = 0; i < 8; ++i) { r[i] = lo[i]; r[i + 8] = hi[i]; }
  return r;
}

__device__ __forceinline__ v8f wmma_bf16(v16bf a, v16bf b, v8f c) {
  return __builtin_amdgcn_wmma_f32_16x16x32_bf16(
      /*neg_a=*/false, a, /*neg_b=*/false, b,
      /*c_mod=*/(short)0, c, /*reuse_a=*/false, /*reuse_b=*/false);
}

// ---------------- conversion kernels ----------------
__global__ void cvt_bf16_kernel(const float* __restrict__ in,
                                __bf16* __restrict__ out, int n) {
  int i = blockIdx.x * blockDim.x + threadIdx.x;
  int stride = gridDim.x * blockDim.x;
  for (; i < n; i += stride) out[i] = (__bf16)in[i];
}

// W[K][N] fp32 -> Wt[N][K] bf16.  Grid: (N/256, K); no integer division.
__global__ void transpose_cvt_kernel(const float* __restrict__ W,
                                     __bf16* __restrict__ Wt, int K, int N) {
  int n = blockIdx.x * blockDim.x + threadIdx.x;
  int k = blockIdx.y;
  if (n < N) Wt[(size_t)n * K + k] = (__bf16)W[(size_t)k * N + n];
}

// ---------------- WMMA GEMM: C[M,N] = A[M,K] * Bt[N,K]^T ----------------
// 128 threads = 4 waves; wave w computes a 32x64 tile (2 M-subtiles x 4
// N-tiles), WG computes 128x64. Ping-pong software pipeline (2 fragment sets
// alternate; no register rotation copies). Requires K % 64 == 0.
// mode 0: bf16 row-major out. mode 1: V transposed-per-head bf16 out.
// mode 2: fp32 out + bias.
__global__ __launch_bounds__(128) void gemm_bf16(
    const __bf16* __restrict__ A, const __bf16* __restrict__ Bt,
    __bf16* __restrict__ outb, float* __restrict__ outf,
    const float* __restrict__ bias, int M, int N, int K, int mode) {
  int wave = threadIdx.x >> 5;
  int lane = threadIdx.x & 31;
  int n0 = blockIdx.x * 64;
  int m0 = blockIdx.y * 128 + wave * 32;
  (void)M;

  v8f zero = {};
  v8f acc[2][4];
#pragma unroll
  for (int s = 0; s < 2; ++s)
#pragma unroll
    for (int t = 0; t < 4; ++t) acc[s][t] = zero;

  const __bf16* arow0 = A + (size_t)m0 * K;
  const __bf16* arow1 = A + (size_t)(m0 + 16) * K;

  // phase-A fragments for kk = 0
  v16bf a0 = load_afrag(arow0, K);
  v16bf a1 = load_afrag(arow1, K);
  v16bf bf[4];
#pragma unroll
  for (int t = 0; t < 4; ++t)
    bf[t] = load_bfrag(Bt + (size_t)(n0 + t * 16) * K, K);

  int kk = 0;
  for (; kk + 64 < K; kk += 64) {
    // load phase-B fragments (kk+32) while phase-A WMMAs run
    v16bf c0 = load_afrag(arow0 + kk + 32, K);
    v16bf c1 = load_afrag(arow1 + kk + 32, K);
    v16bf bg[4];
#pragma unroll
    for (int t = 0; t < 4; ++t)
      bg[t] = load_bfrag(Bt + (size_t)(n0 + t * 16) * K + kk + 32, K);
#pragma unroll
    for (int t = 0; t < 4; ++t) acc[0][t] = wmma_bf16(a0, bf[t], acc[0][t]);
#pragma unroll
    for (int t = 0; t < 4; ++t) acc[1][t] = wmma_bf16(a1, bf[t], acc[1][t]);
    // load next phase-A fragments (kk+64) while phase-B WMMAs run
    a0 = load_afrag(arow0 + kk + 64, K);
    a1 = load_afrag(arow1 + kk + 64, K);
#pragma unroll
    for (int t = 0; t < 4; ++t)
      bf[t] = load_bfrag(Bt + (size_t)(n0 + t * 16) * K + kk + 64, K);
#pragma unroll
    for (int t = 0; t < 4; ++t) acc[0][t] = wmma_bf16(c0, bg[t], acc[0][t]);
#pragma unroll
    for (int t = 0; t < 4; ++t) acc[1][t] = wmma_bf16(c1, bg[t], acc[1][t]);
  }
  // tail: k-steps kk and kk+32
  {
    v16bf c0 = load_afrag(arow0 + kk + 32, K);
    v16bf c1 = load_afrag(arow1 + kk + 32, K);
    v16bf bg[4];
#pragma unroll
    for (int t = 0; t < 4; ++t)
      bg[t] = load_bfrag(Bt + (size_t)(n0 + t * 16) * K + kk + 32, K);
#pragma unroll
    for (int t = 0; t < 4; ++t) acc[0][t] = wmma_bf16(a0, bf[t], acc[0][t]);
#pragma unroll
    for (int t = 0; t < 4; ++t) acc[1][t] = wmma_bf16(a1, bf[t], acc[1][t]);
#pragma unroll
    for (int t = 0; t < 4; ++t) acc[0][t] = wmma_bf16(c0, bg[t], acc[0][t]);
#pragma unroll
    for (int t = 0; t < 4; ++t) acc[1][t] = wmma_bf16(c1, bg[t], acc[1][t]);
  }

  int rlo = (lane < 16) ? 0 : 8;
  int nn = lane & 15;
#pragma unroll
  for (int s = 0; s < 2; ++s) {
#pragma unroll
    for (int t = 0; t < 4; ++t) {
#pragma unroll
      for (int r = 0; r < 8; ++r) {
        int row = m0 + s * 16 + rlo + r;
        int col = n0 + t * 16 + nn;
        float v = acc[s][t][r];
        if (mode == 0) {
          outb[(size_t)row * N + col] = (__bf16)v;
        } else if (mode == 1) {
          int bb = row >> 11, kv = row & (NKV - 1);
          int hh = col >> 6, d = col & (DH - 1);
          outb[(((size_t)(bb * NH + hh) * DH + d) * NKV) + kv] = (__bf16)v;
        } else {
          outf[(size_t)row * N + col] = v + bias[col];
        }
      }
    }
  }
}

// ---------------- fused flash attention ----------------
// Grid: (NQ/64, NH, B). 128 threads = 4 waves; wave w owns 16 q rows.
// Q: [B][NQ][DI] bf16 ; Km: [B][NKV][DI] bf16 ; Vt: [B][NH][DH][NKV] bf16
// H: [B][NQ][DI] bf16 (attention output, pre out-proj)
__global__ __launch_bounds__(128) void attn_kernel(
    const __bf16* __restrict__ Q, const __bf16* __restrict__ Km,
    const __bf16* __restrict__ Vt, __bf16* __restrict__ H) {
  __shared__ __bf16 pshare[4][16][64];  // per-wave 16x64 P tile (bf16)

  int wave = threadIdx.x >> 5;
  int lane = threadIdx.x & 31;
  int qblk = blockIdx.x;
  int h = blockIdx.y;
  int b = blockIdx.z;
  int q0 = qblk * 64 + wave * 16;

  const __bf16* qbase = Q + ((size_t)(b * NQ + q0)) * DI + h * DH;
  v16bf aq0 = load_afrag(qbase, DI);
  v16bf aq1 = load_afrag(qbase + 32, DI);

  const __bf16* kbase = Km + (size_t)b * NKV * DI + h * DH;
  const __bf16* vbase = Vt + ((size_t)(b * NH + h)) * DH * NKV;

  v8f zero = {};
  v8f o[4];
  float m[8], l[8];
#pragma unroll
  for (int t = 0; t < 4; ++t) o[t] = zero;
#pragma unroll
  for (int r = 0; r < 8; ++r) { m[r] = -1e30f; l[r] = 0.f; }

  const float scale = 0.125f;  // 1/sqrt(DH)

  for (int kv0 = 0; kv0 < NKV; kv0 += 64) {
    if (kv0 + 64 < NKV) {  // prefetch next chunk -> global_prefetch_b8
      __builtin_prefetch(kbase + (size_t)(kv0 + 64) * DI, 0, 1);
      __builtin_prefetch(vbase + kv0 + 64, 0, 1);
    }
    // ---- load all K fragments for this chunk, then S-WMMA burst ----
    v16bf kb[4][2];
#pragma unroll
    for (int t = 0; t < 4; ++t) {
      const __bf16* kt = kbase + (size_t)(kv0 + t * 16) * DI;
      kb[t][0] = load_bfrag(kt, DI);
      kb[t][1] = load_bfrag(kt + 32, DI);
    }
    v8f s[4];
#pragma unroll
    for (int t = 0; t < 4; ++t) {
      v8f a = zero;
      a = wmma_bf16(aq0, kb[t][0], a);
      a = wmma_bf16(aq1, kb[t][1], a);
#pragma unroll
      for (int r = 0; r < 8; ++r) a[r] = a[r] * scale;
      s[t] = a;
    }

    // ---- issue V fragment loads now; latency hides under softmax ----
    v16bf vb[4][2];
#pragma unroll
    for (int t = 0; t < 4; ++t) {
      const __bf16* vt = vbase + (size_t)(t * 16) * NKV + kv0;
      vb[t][0] = load_bfrag(vt, NKV);
      vb[t][1] = load_bfrag(vt + 32, NKV);
    }

    // ---- online softmax: row max / rescale ----
    float nm[8], corr[8];
#pragma unroll
    for (int r = 0; r < 8; ++r) {
      float mx = s[0][r];
#pragma unroll
      for (int t = 1; t < 4; ++t) mx = fmaxf(mx, s[t][r]);
#pragma unroll
      for (int off = 1; off < 16; off <<= 1)
        mx = fmaxf(mx, __shfl_xor(mx, off, 16));
      float nmx = fmaxf(m[r], mx);
      corr[r] = __expf(m[r] - nmx);
      nm[r] = nmx;
      m[r] = nmx;
    }
#pragma unroll
    for (int r = 0; r < 8; ++r) {
      float rs = 0.f;
#pragma unroll
      for (int t = 0; t < 4; ++t) {
        float p = __expf(s[t][r] - nm[r]);
        s[t][r] = p;
        rs += p;
      }
#pragma unroll
      for (int off = 1; off < 16; off <<= 1) rs += __shfl_xor(rs, off, 16);
      l[r] = l[r] * corr[r] + rs;
    }
#pragma unroll
    for (int t = 0; t < 4; ++t)
#pragma unroll
      for (int r = 0; r < 8; ++r) o[t][r] = o[t][r] * corr[r];

    // ---- C-layout P -> bf16 A-layout via per-wave LDS round trip ----
    int rlo = (lane < 16) ? 0 : 8;
    int nn = lane & 15;
#pragma unroll
    for (int t = 0; t < 4; ++t)
#pragma unroll
      for (int r = 0; r < 8; ++r)
        pshare[wave][rlo + r][t * 16 + nn] = (__bf16)s[t][r];
    __syncthreads();
    v16bf ap0 = load_afrag(&pshare[wave][0][0], 64);
    v16bf ap1 = load_afrag(&pshare[wave][0][0] + 32, 64);

    // ---- O += P V (V fragments already resident) ----
#pragma unroll
    for (int t = 0; t < 4; ++t) {
      o[t] = wmma_bf16(ap0, vb[t][0], o[t]);
      o[t] = wmma_bf16(ap1, vb[t][1], o[t]);
    }
    __syncthreads();
  }

  // ---- normalize + store bf16 H ----
  int rlo = (lane < 16) ? 0 : 8;
  int nn = lane & 15;
#pragma unroll
  for (int r = 0; r < 8; ++r) {
    float inv = 1.f / l[r];
#pragma unroll
    for (int t = 0; t < 4; ++t) {
      float v = o[t][r] * inv;
      H[((size_t)(b * NQ + q0 + rlo + r)) * DI + h * DH + t * 16 + nn] =
          (__bf16)v;
    }
  }
}

// ---------------- host launcher ----------------
extern "C" void kernel_launch(void* const* d_in, const int* in_sizes, int n_in,
                              void* d_out, int out_size, void* d_ws,
                              size_t ws_size, hipStream_t stream) {
  (void)in_sizes; (void)n_in; (void)out_size; (void)ws_size;
  const float* x   = (const float*)d_in[0];
  const float* ctx = (const float*)d_in[1];
  const float* Wq  = (const float*)d_in[2];
  const float* Wk  = (const float*)d_in[3];
  const float* Wv  = (const float*)d_in[4];
  const float* Wo  = (const float*)d_in[5];
  const float* bo  = (const float*)d_in[6];
  float* out = (float*)d_out;

  char* ws = (char*)d_ws;
  auto alloc = [&](size_t bytes) {
    char* p = ws;
    ws += (bytes + 255) & ~(size_t)255;
    return p;
  };
  const int B = 2;
  const size_t mrows = (size_t)B * NQ;  // 4096
  __bf16* xb  = (__bf16*)alloc(mrows * DX * 2);
  __bf16* cb  = (__bf16*)alloc(mrows * DCTX * 2);
  __bf16* WqT = (__bf16*)alloc((size_t)DI * DX * 2);
  __bf16* WkT = (__bf16*)alloc((size_t)DI * DCTX * 2);
  __bf16* WvT = (__bf16*)alloc((size_t)DI * DCTX * 2);
  __bf16* WoT = (__bf16*)alloc((size_t)DI * DI * 2);
  __bf16* Qb  = (__bf16*)alloc(mrows * DI * 2);
  __bf16* Kb  = (__bf16*)alloc(mrows * DI * 2);
  __bf16* Vtb = (__bf16*)alloc(mrows * DI * 2);
  __bf16* Hb  = (__bf16*)alloc(mrows * DI * 2);

  // 1) convert activations + weights (weights transposed to [N][K])
  cvt_bf16_kernel<<<dim3(2048), dim3(256), 0, stream>>>(x, xb, (int)(mrows * DX));
  cvt_bf16_kernel<<<dim3(2048), dim3(256), 0, stream>>>(ctx, cb, (int)(mrows * DCTX));
  transpose_cvt_kernel<<<dim3(DI / 256, DX), dim3(256), 0, stream>>>(Wq, WqT, DX, DI);
  transpose_cvt_kernel<<<dim3(DI / 256, DCTX), dim3(256), 0, stream>>>(Wk, WkT, DCTX, DI);
  transpose_cvt_kernel<<<dim3(DI / 256, DCTX), dim3(256), 0, stream>>>(Wv, WvT, DCTX, DI);
  transpose_cvt_kernel<<<dim3(DI / 256, DI), dim3(256), 0, stream>>>(Wo, WoT, DI, DI);

  // 2) projections (Q, K row-major; V transposed-per-head)
  gemm_bf16<<<dim3(DI / 64, mrows / 128), dim3(128), 0, stream>>>(
      xb, WqT, Qb, nullptr, nullptr, (int)mrows, DI, DX, 0);
  gemm_bf16<<<dim3(DI / 64, mrows / 128), dim3(128), 0, stream>>>(
      cb, WkT, Kb, nullptr, nullptr, (int)mrows, DI, DCTX, 0);
  gemm_bf16<<<dim3(DI / 64, mrows / 128), dim3(128), 0, stream>>>(
      cb, WvT, Vtb, nullptr, nullptr, (int)mrows, DI, DCTX, 1);

  // 3) fused flash attention
  attn_kernel<<<dim3(NQ / 64, NH, B), dim3(128), 0, stream>>>(Qb, Kb, Vtb, Hb);

  // 4) output projection + bias (fp32 out)
  gemm_bf16<<<dim3(DI / 64, mrows / 128), dim3(128), 0, stream>>>(
      Hb, WoT, nullptr, out, bo, (int)mrows, DI, DI, 2);
}
